// GATLayer_17635135717521
// MI455X (gfx1250) — matrix operand order repaired and checked by reading, not compiled, not run on hardware
//
#include <hip/hip_runtime.h>
#include <hip/hip_bf16.h>

typedef __attribute__((ext_vector_type(16))) _Float16 v16h;
typedef __attribute__((ext_vector_type(8)))  float    v8f;

#define DIMK   64     // feature dim per head
#define HEADS  4
#define FDIM   256    // HEADS * DIMK

// ---- order-preserving float <-> uint encoding for atomicMax-based segment max ----
__device__ __forceinline__ unsigned f2ord(float f) {
    unsigned u = __float_as_uint(f);
    return (u & 0x80000000u) ? ~u : (u | 0x80000000u);
}
__device__ __forceinline__ float ord2f(unsigned e) {
    return (e & 0x80000000u) ? __uint_as_float(e & 0x7fffffffu) : __uint_as_float(~e);
}
#define ORD_NEG_INF 0x007FFFFFu   // f2ord(-inf)

// ---------------------------------------------------------------------------
// Init workspace: agg = 0, denom = 0, smax = enc(-inf)
// ---------------------------------------------------------------------------
__global__ void gat_init(float* __restrict__ agg, float* __restrict__ denom,
                         unsigned* __restrict__ smax, int N) {
    long long idx = (long long)blockIdx.x * blockDim.x + threadIdx.x;
    long long na  = (long long)N * FDIM;
    if (idx < na)               agg[idx] = 0.0f;
    else if (idx < na + N)      denom[idx - na] = 0.0f;
    else if (idx < na + 2LL*N)  smax[idx - na - N] = ORD_NEG_INF;
}

// ---------------------------------------------------------------------------
// ft[N,256] = h_v[N,64] @ W_fc[256,64]^T + b_fc  -- one wave per 16x16 tile,
// K=64 split into two v_wmma_f32_16x16x32_f16 steps (f32 accumulate).
// A layout (16-bit, 16x32): lane l holds row M=l&15; K groups of 8 selected
// by l>>4 (K = ko..ko+7 then ko+16..ko+23). B layout: lane l holds column
// N=l&15, same split K groups of 8 as A. C/D: VGPR j -> row j + 8*(l>>4).
// Epilogue: single tile-level guard; full tiles take an unguarded 8-store
// path (N % 16 == 0 in practice, so the branchy tail never runs).
// ---------------------------------------------------------------------------
__global__ void gat_gemm_ft(const float* __restrict__ hv,
                            const float* __restrict__ Wfc,
                            const float* __restrict__ bfc,
                            float* __restrict__ ft, int N, int nTiles) {
    int tile = blockIdx.x * (blockDim.x >> 5) + (threadIdx.x >> 5);
    if (tile >= nTiles) return;
    int lane = threadIdx.x & 31;
    int tcol = tile & 15;          // FDIM/16 = 16 column tiles
    int trow = tile >> 4;
    int m    = lane & 15;
    int half = lane >> 4;

    int row  = trow * 16 + m;
    int rowc = row < N ? row : N - 1;     // clamp tail-row loads
    int col  = tcol * 16 + m;             // B column / D column

    v8f acc = {};
    #pragma unroll
    for (int ks = 0; ks < 2; ++ks) {
        v16h a, b;
        const float* ap = hv + (size_t)rowc * DIMK + ks * 32 + half * 8;
        #pragma unroll
        for (int i = 0; i < 8; ++i) a[i]     = (_Float16)ap[i];
        #pragma unroll
        for (int i = 0; i < 8; ++i) a[8 + i] = (_Float16)ap[16 + i];
        const float* bp = Wfc + (size_t)col * DIMK + ks * 32 + half * 8;
        #pragma unroll
        for (int i = 0; i < 8; ++i) b[i]     = (_Float16)bp[i];
        #pragma unroll
        for (int i = 0; i < 8; ++i) b[8 + i] = (_Float16)bp[16 + i];
        acc = __builtin_amdgcn_wmma_f32_16x16x32_f16(false, a, false, b,
                                                     (short)0, acc, false, false);
    }
    float bias = bfc[col];
    int   mr0  = trow * 16 + half * 8;
    float* dp  = ft + (size_t)mr0 * FDIM + col;
    if (mr0 + 8 <= N) {                       // full tile: unguarded stores
        #pragma unroll
        for (int j = 0; j < 8; ++j) dp[(size_t)j * FDIM] = acc[j] + bias;
    } else {                                  // tail tile (never hit for N%16==0)
        #pragma unroll
        for (int j = 0; j < 8; ++j)
            if (mr0 + j < N) dp[(size_t)j * FDIM] = acc[j] + bias;
    }
}

// ---------------------------------------------------------------------------
// Per-edge score: s = leakyrelu_0.2( sum_j W_pi[j]*ft[src][j]*ft[dst][j] )
// One wave per edge; rows of ft are L2-resident (51 MB < 192 MB L2).
// Segment max via atomicMax on order-preserving uint encoding.
// ---------------------------------------------------------------------------
__global__ void gat_edge_score(const float* __restrict__ ft,
                               const float* __restrict__ Wpi,
                               const int* __restrict__ src,
                               const int* __restrict__ dst,
                               float* __restrict__ s_out,
                               unsigned* __restrict__ smax_bits, int E) {
    long long gid = (long long)blockIdx.x * blockDim.x + threadIdx.x;
    int e    = (int)(gid >> 5);
    int lane = (int)(gid & 31);
    if (e >= E) return;

    const float4* fs = (const float4*)(ft + (size_t)src[e] * FDIM) + lane * 2;
    const float4* fd = (const float4*)(ft + (size_t)dst[e] * FDIM) + lane * 2;
    const float4* fw = (const float4*)Wpi + lane * 2;
    float4 s0 = fs[0], s1 = fs[1];
    float4 d0 = fd[0], d1 = fd[1];
    float4 w0 = fw[0], w1 = fw[1];

    float p = w0.x*s0.x*d0.x + w0.y*s0.y*d0.y + w0.z*s0.z*d0.z + w0.w*s0.w*d0.w
            + w1.x*s1.x*d1.x + w1.y*s1.y*d1.y + w1.z*s1.z*d1.z + w1.w*s1.w*d1.w;
    #pragma unroll
    for (int off = 16; off > 0; off >>= 1) p += __shfl_xor(p, off, 32);

    if (lane == 0) {
        float sv = p > 0.0f ? p : 0.2f * p;
        s_out[e] = sv;
        atomicMax(&smax_bits[dst[e]], f2ord(sv));
    }
}

// ---------------------------------------------------------------------------
// ex = exp(s - smax[dst]); denom[dst] += ex   (one thread per edge)
// ---------------------------------------------------------------------------
__global__ void gat_edge_exp(const float* __restrict__ s_in,
                             const int* __restrict__ dst,
                             const unsigned* __restrict__ smax_bits,
                             float* __restrict__ ex_out,
                             float* __restrict__ denom, int E) {
    int e = blockIdx.x * blockDim.x + threadIdx.x;
    if (e >= E) return;
    int d = dst[e];
    float sm = ord2f(smax_bits[d]);
    float ex = __expf(s_in[e] - sm);
    ex_out[e] = ex;
    unsafeAtomicAdd(&denom[d], ex);
}

// ---------------------------------------------------------------------------
// agg[dst] += ft[src] * (ex/denom[dst])  -- one wave per edge, 8 f32 atomic
// adds per lane (global_atomic_add_f32, resolved in L2).
// ---------------------------------------------------------------------------
__global__ void gat_edge_agg(const float* __restrict__ ft,
                             const int* __restrict__ src,
                             const int* __restrict__ dst,
                             const float* __restrict__ ex,
                             const float* __restrict__ denom,
                             float* __restrict__ agg, int E) {
    long long gid = (long long)blockIdx.x * blockDim.x + threadIdx.x;
    int e    = (int)(gid >> 5);
    int lane = (int)(gid & 31);
    if (e >= E) return;

    int d = dst[e];
    float a = ex[e] / denom[d];
    const float4* fs = (const float4*)(ft + (size_t)src[e] * FDIM) + lane * 2;
    float4 s0 = fs[0], s1 = fs[1];
    float* ag = agg + (size_t)d * FDIM + lane * 8;
    unsafeAtomicAdd(ag + 0, s0.x * a);
    unsafeAtomicAdd(ag + 1, s0.y * a);
    unsafeAtomicAdd(ag + 2, s0.z * a);
    unsafeAtomicAdd(ag + 3, s0.w * a);
    unsafeAtomicAdd(ag + 4, s1.x * a);
    unsafeAtomicAdd(ag + 5, s1.y * a);
    unsafeAtomicAdd(ag + 6, s1.z * a);
    unsafeAtomicAdd(ag + 7, s1.w * a);
}

// ---------------------------------------------------------------------------
// out[n,d] = max over 4 heads of agg[n, h*64+d]
// ---------------------------------------------------------------------------
__global__ void gat_pool(const float* __restrict__ agg,
                         float* __restrict__ out, int N) {
    int idx = blockIdx.x * blockDim.x + threadIdx.x;
    if (idx >= N * DIMK) return;
    int n = idx >> 6, d = idx & 63;
    const float* base = agg + (size_t)n * FDIM + d;
    float m = base[0];
    m = fmaxf(m, base[64]);
    m = fmaxf(m, base[128]);
    m = fmaxf(m, base[192]);
    out[idx] = m;
}

// ---------------------------------------------------------------------------
extern "C" void kernel_launch(void* const* d_in, const int* in_sizes, int n_in,
                              void* d_out, int out_size, void* d_ws, size_t ws_size,
                              hipStream_t stream) {
    const float* h_v = (const float*)d_in[0];
    const int*   src = (const int*)d_in[1];
    const int*   dst = (const int*)d_in[2];
    const float* Wfc = (const float*)d_in[3];
    const float* bfc = (const float*)d_in[4];
    const float* Wpi = (const float*)d_in[5];
    int N = in_sizes[0] / DIMK;
    int E = in_sizes[1];
    float* out = (float*)d_out;

    char* p = (char*)d_ws;
    float*    ft    = (float*)p;    p += (size_t)N * FDIM * sizeof(float);
    float*    agg   = (float*)p;    p += (size_t)N * FDIM * sizeof(float);
    float*    s     = (float*)p;    p += (size_t)E * sizeof(float);
    float*    ex    = (float*)p;    p += (size_t)E * sizeof(float);
    float*    denom = (float*)p;    p += (size_t)N * sizeof(float);
    unsigned* smax  = (unsigned*)p; p += (size_t)N * sizeof(unsigned);

    long long initN = (long long)N * FDIM + 2LL * N;
    gat_init<<<(unsigned)((initN + 255) / 256), 256, 0, stream>>>(agg, denom, smax, N);

    int nTiles = ((N + 15) / 16) * 16;   // row-tiles * 16 column-tiles
    gat_gemm_ft<<<(unsigned)((nTiles + 7) / 8), 256, 0, stream>>>(h_v, Wfc, bfc, ft, N, nTiles);

    long long tE = (long long)E * 32;
    gat_edge_score<<<(unsigned)((tE + 255) / 256), 256, 0, stream>>>(ft, Wpi, src, dst, s, smax, E);
    gat_edge_exp<<<(unsigned)((E + 255) / 256), 256, 0, stream>>>(s, dst, smax, ex, denom, E);
    gat_edge_agg<<<(unsigned)((tE + 255) / 256), 256, 0, stream>>>(ft, src, dst, ex, denom, agg, E);
    gat_pool<<<(unsigned)(((long long)N * DIMK + 255) / 256), 256, 0, stream>>>(agg, out, N);
}
